// WeaklySupervisedSegmentationModelUNet_70712341561529
// MI455X (gfx1250) — matrix operand order repaired and checked by reading, not compiled, not run on hardware
//
#include <hip/hip_runtime.h>

#define Bn 8
#define Cn 21
#define Hn 256
#define Wn 256
#define HWn 65536
#define Dn 64
#define CAM_TH 0.1f
#define EPSf 1e-8f
#define NTILES 4096   // HWn / 16 pixel-tiles per batch

typedef __attribute__((ext_vector_type(16))) _Float16 v16h;
typedef __attribute__((ext_vector_type(8)))  float    v8f;

// ---------------------------------------------------------------------------
// K1: pooled[b,c] = mean(relu(cam)); refined init = (relu(cam) > 0.1)
// one block per (b,c)
// ---------------------------------------------------------------------------
__global__ void pool_init_kernel(const float* __restrict__ cam_raw,
                                 float* __restrict__ pooled,
                                 float* __restrict__ refined) {
  int bc = blockIdx.x, tid = threadIdx.x;
  const float* src = cam_raw + (size_t)bc * HWn;
  float* dst = refined + (size_t)bc * HWn;
  float s = 0.f;
  for (int p = tid; p < HWn; p += 256) {
    float v = src[p]; v = v > 0.f ? v : 0.f;
    dst[p] = (v > CAM_TH) ? 1.f : 0.f;
    s += v;
  }
  __shared__ float red[256];
  red[tid] = s; __syncthreads();
  for (int off = 128; off > 0; off >>= 1) {
    if (tid < off) red[tid] += red[tid + off];
    __syncthreads();
  }
  if (tid == 0) pooled[bc] = red[0] * (1.f / (float)HWn);
}

// ---------------------------------------------------------------------------
// K2: projection GEMM (16pix x 21ch)x(21ch x 64d) via WMMA f16 (f32 acc),
// L2-normalize over d, then emit normed in TWO layouts:
//   planar  [B,D,HW]   f16  -> coalesced stride-1 reads in fgsum_kernel
//   frag    [tile,lane,32h] f16 -> exact WMMA A-operand fragments for sim
// block = 128 threads (4 waves), wave = one 16-pixel tile.
// ---------------------------------------------------------------------------
__global__ void project_kernel(const float* __restrict__ cam_raw,
                               const float* __restrict__ proj_w,
                               const float* __restrict__ proj_b,
                               _Float16* __restrict__ normed,
                               _Float16* __restrict__ frag) {
  __shared__ float lfeat[4][16][Dn + 1];
  __shared__ float rnArr[64];
  int tid = threadIdx.x;
  int wave = tid >> 5, lane = tid & 31;
  size_t pixBase = (size_t)blockIdx.x * 64;
  int b  = (int)(pixBase / HWn);
  int pb = (int)(pixBase % HWn);
  int p0 = pb + wave * 16;
  int m = lane & 15, half = lane >> 4;
  int kbase = half * 8;
  int pix = p0 + m;
  const float* camb = cam_raw + (size_t)b * Cn * HWn;

  // A operand: 16x32 f16, K = channel (21 valid, zero-pad via select — no
  // divergent exec-mask branches: clamp the address, cndmask the value).
  v16h a;
#pragma unroll
  for (int i = 0; i < 16; ++i) {
    int c = kbase + i + ((i >= 8) ? 8 : 0);
    int cc = (c < Cn) ? c : 0;
    float v = camb[(size_t)cc * HWn + pix];
    v = v > 0.f ? v : 0.f;
    a[i] = (c < Cn) ? (_Float16)v : (_Float16)0.f;
  }

#pragma unroll
  for (int nt = 0; nt < 4; ++nt) {
    int d = nt * 16 + m;              // N index for this lane
    v16h bm;
#pragma unroll
    for (int i = 0; i < 16; ++i) {
      int c = half * 8 + i + ((i >= 8) ? 8 : 0);          // K index
      int cc = (c < Cn) ? c : 0;
      float wv = proj_w[d * Cn + cc];
      bm[i] = (c < Cn) ? (_Float16)wv : (_Float16)0.f;
    }
    v8f acc = {};
    acc = __builtin_amdgcn_wmma_f32_16x16x32_f16(false, a, false, bm,
                                                 (short)0, acc, false, false);
    float bias = proj_b[d];
#pragma unroll
    for (int r = 0; r < 8; ++r)
      lfeat[wave][r + 8 * half][d] = acc[r] + bias;   // M = r + 8*half
  }
  __syncthreads();

  // ---- normalize: thread pair (pixel = t>>1, D-half = t&1) ----
  int pl = tid >> 1, dh = tid & 1;
  int w2 = pl >> 4, m2 = pl & 15;
  float ss = 0.f;
#pragma unroll
  for (int d = dh * 32; d < dh * 32 + 32; ++d) {
    float f = lfeat[w2][m2][d]; ss += f * f;
  }
  ss += __shfl_xor(ss, 1);
  float rn = 1.f / (sqrtf(ss) + EPSf);
  if (dh == 0) rnArr[pl] = rn;

  // ---- fragment-layout writes: this thread owns d-chunk q=dh of pixel pl.
  // frag element (tileG, lane, j): lane = h*16 + m, j = q*16 + i,
//   d = q*32 + h*16 + i   (matches WMMA A layout in sim)
  size_t tileG = (size_t)blockIdx.x * 4 + w2;
#pragma unroll
  for (int h = 0; h < 2; ++h) {
    v16h fv;
#pragma unroll
    for (int i = 0; i < 16; ++i) {
      int d = dh * 32 + h * 16 + i;
      fv[i] = (_Float16)(lfeat[w2][m2][d] * rn);
    }
    *(v16h*)(frag + ((tileG * 32 + (h * 16 + m2)) * 32 + dh * 16)) = fv;
  }
  __syncthreads();

  // ---- planar writes, coalesced: consecutive threads -> consecutive pixels
  int px = tid & 63;        // pixel in block
  int ds = tid >> 6;        // 0 or 1
  int w3 = px >> 4, m3 = px & 15;
  float rn2 = rnArr[px];
  _Float16* nb = normed + (size_t)b * Dn * HWn + pb + px;
  for (int d = ds; d < Dn; d += 2)
    nb[(size_t)d * HWn] = (_Float16)(lfeat[w3][m3][d] * rn2);
}

// ---------------------------------------------------------------------------
// K4b: fg_cnt[b,c] = #(refined > 0.5); one block per (b,c)
// ---------------------------------------------------------------------------
__global__ void count_kernel(const float* __restrict__ refined,
                             float* __restrict__ fg_cnt) {
  int bc = blockIdx.x, tid = threadIdx.x;
  const float* r = refined + (size_t)bc * HWn;
  float s = 0.f;
  for (int p = tid; p < HWn; p += 256) s += (r[p] > 0.5f) ? 1.f : 0.f;
  __shared__ float red[256];
  red[tid] = s; __syncthreads();
  for (int off = 128; off > 0; off >>= 1) {
    if (tid < off) red[tid] += red[tid + off];
    __syncthreads();
  }
  if (tid == 0) fg_cnt[bc] = red[0];
}

// ---------------------------------------------------------------------------
// K4a: fg_partial[b,chunk,c,d] = sum over 1024-pixel chunk of fg*normed.
// Two-stage (deterministic, no atomics). grid = (64 chunks, B)
// ---------------------------------------------------------------------------
__global__ void fgsum_kernel(const float* __restrict__ refined,
                             const _Float16* __restrict__ normed,
                             float* __restrict__ fg_partial) {
  __shared__ unsigned msk[1024];
  int tid = threadIdx.x;
  int b = blockIdx.y, chunk = blockIdx.x;
  int base = chunk * 1024;
  const float* rb = refined + (size_t)b * Cn * HWn;
  for (int j = 0; j < 4; ++j) {
    int pl = j * 256 + tid;
    unsigned mm = 0;
#pragma unroll
    for (int c = 0; c < Cn; ++c)
      mm |= (rb[(size_t)c * HWn + base + pl] > 0.5f) ? (1u << c) : 0u;
    msk[pl] = mm;
  }
  __syncthreads();
  int d = tid >> 2, ps = tid & 3;   // 4 threads share one d
  float acc[Cn];
#pragma unroll
  for (int c = 0; c < Cn; ++c) acc[c] = 0.f;
  const _Float16* nd = normed + ((size_t)b * Dn + d) * HWn + base;
  for (int j = 0; j < 256; ++j) {
    int pl = j * 4 + ps;
    unsigned mm = msk[pl];
    if (!mm) continue;
    float v = (float)nd[pl];
#pragma unroll
    for (int c = 0; c < Cn; ++c) acc[c] += (mm & (1u << c)) ? v : 0.f;
  }
#pragma unroll
  for (int c = 0; c < Cn; ++c) {
    acc[c] += __shfl_xor(acc[c], 1);
    acc[c] += __shfl_xor(acc[c], 2);
  }
  if (ps == 0) {
    float* out = fg_partial + (((size_t)b * 64 + chunk) * Cn) * Dn + d;
#pragma unroll
    for (int c = 0; c < Cn; ++c) out[(size_t)c * Dn] = acc[c];
  }
}

// ---------------------------------------------------------------------------
// K5: fg_mean (f16) + active flag. one block per (b,c), 64 threads (d)
// ---------------------------------------------------------------------------
__global__ void mean_kernel(const float* __restrict__ fg_partial,
                            const float* __restrict__ fg_cnt,
                            const int* __restrict__ labels,
                            _Float16* __restrict__ fg_mean,
                            float* __restrict__ active) {
  int bc = blockIdx.x, d = threadIdx.x;
  int b = bc / Cn, c = bc % Cn;
  float s = 0.f;
  for (int ch = 0; ch < 64; ++ch)
    s += fg_partial[(((size_t)b * 64 + ch) * Cn + c) * Dn + d];
  float cnt = fg_cnt[bc];
  fg_mean[(size_t)bc * Dn + d] = (_Float16)(s / (cnt + EPSf));
  if (d == 0) active[bc] = (cnt > 0.f && labels[bc] > 0) ? 1.f : 0.f;
}

// ---------------------------------------------------------------------------
// K6: sim[b,c,p] = <normed[b,:,p], fg_mean[b,c,:]>  (GEMM K=64, 2x WMMA f16)
// A-operands come pre-swizzled from `frag` (2x 32-byte coalesced loads/lane),
// B-operands are contiguous v16h LDS loads, D stored as one v8f vector.
// ---------------------------------------------------------------------------
__global__ void sim_kernel(const _Float16* __restrict__ frag,
                           const _Float16* __restrict__ fg_mean,
                           float* __restrict__ sim) {
  __shared__ __align__(64) _Float16 lmean[32][Dn];  // c padded to 32, zeros
  int tid = threadIdx.x;
  size_t pixBase = (size_t)blockIdx.x * 64;
  int b  = (int)(pixBase / HWn);
  int pb = (int)(pixBase % HWn);
  for (int k = tid; k < 32 * Dn; k += 128) {
    int c = k >> 6, d = k & 63;
    lmean[c][d] = (c < Cn) ? fg_mean[((size_t)b * Cn + c) * Dn + d]
                           : (_Float16)0.f;
  }
  __syncthreads();

  int wave = tid >> 5, lane = tid & 31;
  int p0 = pb + wave * 16;
  int m = lane & 15, half = lane >> 4;

  // A fragments: 64 contiguous bytes per lane
  const v16h* af =
      (const v16h*)(frag + ((size_t)(blockIdx.x * 4 + wave) * 32 + lane) * 32);
  v16h a0 = af[0];   // K = d in [0,32)
  v16h a1 = af[1];   // K = d in [32,64)

#pragma unroll
  for (int nt = 0; nt < 2; ++nt) {
    int c = nt * 16 + m;              // N index
    // B layout: N = lane&15, K = 16*half + i  -> 32 contiguous bytes
    v16h b0 = *(const v16h*)&lmean[c][half * 16];
    v16h b1 = *(const v16h*)&lmean[c][half * 16 + 32];
    v8f acc = {};
    acc = __builtin_amdgcn_wmma_f32_16x16x32_f16(false, a0, false, b0,
                                                 (short)0, acc, false, false);
    acc = __builtin_amdgcn_wmma_f32_16x16x32_f16(false, a1, false, b1,
                                                 (short)0, acc, false, false);
    if (c < Cn) {
      // M = r + 8*half -> 8 consecutive pixels: one 32B vector store
      *(v8f*)(sim + ((size_t)b * Cn + c) * HWn + p0 + 8 * half) = acc;
    }
  }
}

// ---------------------------------------------------------------------------
// K7: per-(b,c) mean/std(ddof=1)/max of sim -> threshold + max
// ---------------------------------------------------------------------------
__global__ void simstats_kernel(const float* __restrict__ sim, float strict,
                                float* __restrict__ thrArr,
                                float* __restrict__ maxArr) {
  int bc = blockIdx.x, tid = threadIdx.x;
  const float* s = sim + (size_t)bc * HWn;
  float sum = 0.f, sq = 0.f, mx = -3.4e38f;
  for (int p = tid; p < HWn; p += 256) {
    float v = s[p]; sum += v; sq += v * v; mx = v > mx ? v : mx;
  }
  __shared__ float r1[256], r2[256], r3[256];
  r1[tid] = sum; r2[tid] = sq; r3[tid] = mx; __syncthreads();
  for (int off = 128; off > 0; off >>= 1) {
    if (tid < off) {
      r1[tid] += r1[tid + off];
      r2[tid] += r2[tid + off];
      r3[tid] = fmaxf(r3[tid], r3[tid + off]);
    }
    __syncthreads();
  }
  if (tid == 0) {
    float N = (float)HWn;
    float mean = r1[0] / N;
    float var = (r2[0] - r1[0] * r1[0] / N) / (N - 1.f);
    var = var > 0.f ? var : 0.f;
    thrArr[bc] = mean + strict * sqrtf(var);
    maxArr[bc] = r3[0];
  }
}

// ---------------------------------------------------------------------------
// K8: region-grow update (ping-pong so dilation reads a consistent snapshot)
// ---------------------------------------------------------------------------
__global__ void grow_kernel(const float* __restrict__ src,
                            float* __restrict__ dst,
                            const float* __restrict__ sim,
                            const float* __restrict__ thrArr,
                            const float* __restrict__ maxArr,
                            const float* __restrict__ active, int it) {
  size_t idx = (size_t)blockIdx.x * 256 + threadIdx.x;
  int bc = (int)(idx >> 16);
  int p  = (int)(idx & 65535);
  float rv = src[idx];
  bool fg = rv > 0.5f;
  float sv = sim[idx];
  float th = thrArr[bc];
  bool np = (sv > th) && !fg;
  if (np && it > 0) {
    int h = p >> 8, w = p & 255;
    bool dil = false;
    const float* sb = src + ((size_t)bc << 16);
    for (int dh = -1; dh <= 1; ++dh) {
      int hh = h + dh; if (hh < 0 || hh >= Hn) continue;
      for (int dw = -1; dw <= 1; ++dw) {
        int ww = w + dw; if (ww < 0 || ww >= Wn) continue;
        dil = dil || (sb[hh * 256 + ww] > 0.5f);
      }
    }
    np = dil;
  }
  float conf = (sv - th) / (maxArr[bc] - th + EPSf);
  conf = fminf(fmaxf(conf, 0.1f), 0.9f);
  float add = (np && active[bc] > 0.f) ? conf : 0.f;
  dst[idx] = rv + add;
}

// ---------------------------------------------------------------------------
// K9: clip, first-max argmax over C, one-hot multiply
// ---------------------------------------------------------------------------
__global__ void final_kernel(const float* __restrict__ refined,
                             float* __restrict__ seg) {
  size_t idx = (size_t)blockIdx.x * 256 + threadIdx.x;  // b*HW + p
  int b = (int)(idx >> 16);
  int p = (int)(idx & 65535);
  const float* rb = refined + (size_t)b * Cn * HWn + p;
  float best = -1.f; int bi = 0;
  float vals[Cn];
#pragma unroll
  for (int c = 0; c < Cn; ++c) {
    float v = rb[(size_t)c * HWn];
    v = fminf(fmaxf(v, 0.f), 1.f);
    vals[c] = v;
    if (v > best) { best = v; bi = c; }   // strict > keeps FIRST max
  }
  float* ob = seg + (size_t)b * Cn * HWn + p;
#pragma unroll
  for (int c = 0; c < Cn; ++c) ob[(size_t)c * HWn] = (c == bi) ? vals[c] : 0.f;
}

// ---------------------------------------------------------------------------
extern "C" void kernel_launch(void* const* d_in, const int* in_sizes, int n_in,
                              void* d_out, int out_size, void* d_ws,
                              size_t ws_size, hipStream_t stream) {
  (void)in_sizes; (void)n_in; (void)out_size; (void)ws_size;
  const float* cam_raw = (const float*)d_in[0];
  const float* proj_w  = (const float*)d_in[1];
  const float* proj_b  = (const float*)d_in[2];
  const int*   labels  = (const int*)d_in[3];
  float* pooled = (float*)d_out;                  // [B,C]
  float* seg    = (float*)d_out + Bn * Cn;        // [B,C,H,W]

  char* ws = (char*)d_ws;
  size_t off = 0;
  auto alloc = [&](size_t bytes) -> void* {
    void* p = ws + off;
    off += (bytes + 255) & ~(size_t)255;
    return p;
  };
  _Float16* normed  = (_Float16*)alloc((size_t)Bn * Dn * HWn * 2);      // 67 MB
  _Float16* frag    = (_Float16*)alloc((size_t)Bn * NTILES * 32 * 32 * 2); // 67 MB
  float* refinedA   = (float*)alloc((size_t)Bn * Cn * HWn * 4);         // 44 MB
  float* refinedB   = (float*)alloc((size_t)Bn * Cn * HWn * 4);         // 44 MB
  float* sim        = (float*)alloc((size_t)Bn * Cn * HWn * 4);         // 44 MB
  float* fg_partial = (float*)alloc((size_t)Bn * 64 * Cn * Dn * 4);     // 2.7 MB
  float* fg_cnt     = (float*)alloc((size_t)Bn * Cn * 4);
  _Float16* fg_mean = (_Float16*)alloc((size_t)Bn * Cn * Dn * 2);
  float* thrArr     = (float*)alloc((size_t)Bn * Cn * 4);
  float* maxArr     = (float*)alloc((size_t)Bn * Cn * 4);
  float* activeArr  = (float*)alloc((size_t)Bn * Cn * 4);

  pool_init_kernel<<<Bn * Cn, 256, 0, stream>>>(cam_raw, pooled, refinedA);
  project_kernel<<<(Bn * HWn) / 64, 128, 0, stream>>>(cam_raw, proj_w, proj_b,
                                                      normed, frag);
  for (int it = 0; it < 5; ++it) {
    const float* src = (it & 1) ? refinedB : refinedA;
    float* dst       = (it & 1) ? refinedA : refinedB;
    count_kernel<<<Bn * Cn, 256, 0, stream>>>(src, fg_cnt);
    fgsum_kernel<<<dim3(64, Bn), 256, 0, stream>>>(src, normed, fg_partial);
    mean_kernel<<<Bn * Cn, 64, 0, stream>>>(fg_partial, fg_cnt, labels,
                                            fg_mean, activeArr);
    sim_kernel<<<(Bn * HWn) / 64, 128, 0, stream>>>(frag, fg_mean, sim);
    float strict = 0.5f - 0.05f * (float)it;
    simstats_kernel<<<Bn * Cn, 256, 0, stream>>>(sim, strict, thrArr, maxArr);
    grow_kernel<<<(Bn * Cn * HWn) / 256, 256, 0, stream>>>(src, dst, sim,
                                                           thrArr, maxArr,
                                                           activeArr, it);
  }
  // after 5 iterations the final refined lives in refinedB
  final_kernel<<<(Bn * HWn) / 256, 256, 0, stream>>>(refinedB, seg);
}
